// NeuralMemoryNetwork_51049981281184
// MI455X (gfx1250) — compile-verified, matrix operand order checked by reading, not run on hardware
//
#include <hip/hip_runtime.h>

typedef __attribute__((ext_vector_type(16))) __bf16 v16bf;
typedef __attribute__((ext_vector_type(8)))  __bf16 v8bf;
typedef __attribute__((ext_vector_type(8)))  float  v8f;

constexpr int ROWS = 8192;   // B*S = 4*2048 query rows
constexpr int IND  = 1024;   // IN_DIM
constexpr int MD   = 512;    // MEM_DIM
constexpr int MS   = 8192;   // MEM_SIZE

// ---------- helpers ----------

__device__ __forceinline__ __bf16 f2bf(float f) {
  unsigned int u = __builtin_bit_cast(unsigned int, f);
  unsigned int r = (u + 0x7FFFu + ((u >> 16) & 1u)) >> 16;  // RNE
  unsigned short h = (unsigned short)r;
  return __builtin_bit_cast(__bf16, h);
}

__device__ __forceinline__ v8f wmma_bf16(v16bf a, v16bf b, v8f c) {
  // v_wmma_f32_16x16x32_bf16: D = A(16x32) * B(32x16) + C(16x16 f32)
  return __builtin_amdgcn_wmma_f32_16x16x32_bf16(
      /*neg_a=*/false, a, /*neg_b=*/false, b,
      /*c_mod=*/(short)0, c, /*reuse_a=*/false, /*reuse_b=*/false);
}

// A/B fragment for an "NT" GEMM operand: row-major bf16 matrix, K contiguous.
// ISA 7.12.2: lane%16 = row, lane/16 picks the K octet; VGPR0-3 = K[kh..kh+7],
// VGPR4-7 = K[kh+16..kh+23]. Two 16-byte loads per fragment.
__device__ __forceinline__ v16bf load_frag_bf(const __bf16* base, int ld,
                                              int rowBase, int kBase, int lane) {
  int r  = rowBase + (lane & 15);
  int kh = kBase + ((lane >> 4) << 3);
  const __bf16* p = base + (size_t)r * ld + kh;
  v8bf lo = *(const v8bf*)(p);
  v8bf hi = *(const v8bf*)(p + 16);
  v16bf f;
#pragma unroll
  for (int i = 0; i < 8; ++i) { f[i] = lo[i]; f[8 + i] = hi[i]; }
  return f;
}

// ---------- kernel 0: generic f32 -> bf16 (8 elems/thread, vectorized) ----------

__global__ __launch_bounds__(256)
void k_cvt_bf16(const float* __restrict__ src, __bf16* __restrict__ dst, int n8) {
  int i = blockIdx.x * 256 + threadIdx.x;
  if (i >= n8) return;
  const float4* p = (const float4*)src + 2 * (size_t)i;
  float4 a = p[0], b = p[1];
  v8bf o;
  o[0] = f2bf(a.x); o[1] = f2bf(a.y); o[2] = f2bf(a.z); o[3] = f2bf(a.w);
  o[4] = f2bf(b.x); o[5] = f2bf(b.y); o[6] = f2bf(b.z); o[7] = f2bf(b.w);
  ((v8bf*)dst)[i] = o;
}

// ---------- kernel 1: memory -> bf16 (row-major + transposed) ----------

__global__ __launch_bounds__(256)
void k_memconv(const float* __restrict__ mem, __bf16* __restrict__ mb,
               __bf16* __restrict__ mTb) {
  int idx = blockIdx.x * 256 + threadIdx.x;          // 0 .. MS*MD-1
  int m = idx >> 9;                                  // / MD
  int d = idx & (MD - 1);
  __bf16 v = f2bf(mem[idx]);
  mb[idx] = v;
  mTb[(size_t)d * MS + m] = v;
}

// ---------- kernel 2: enc = x @ W_in^T + b_in  (8192x512, K=1024) ----------
// 8 waves, block tile 64 rows x 128 cols; wave tile 16x64. Pure bf16 operands.

__global__ __launch_bounds__(256)
void k_encode(const __bf16* __restrict__ xb, const __bf16* __restrict__ Winb,
              const float* __restrict__ bin, float* __restrict__ encf,
              __bf16* __restrict__ encb) {
  int lane = threadIdx.x & 31;
  int wave = threadIdx.x >> 5;
  int rowBase = blockIdx.x * 64 + (wave & 3) * 16;
  int colBase = blockIdx.y * 128 + (wave >> 2) * 64;
  v8f acc[4] = {};
  for (int k = 0; k < IND; k += 32) {
    v16bf a = load_frag_bf(xb, IND, rowBase, k, lane);
#pragma unroll
    for (int t = 0; t < 4; ++t) {
      v16bf b = load_frag_bf(Winb, IND, colBase + 16 * t, k, lane);
      acc[t] = wmma_bf16(a, b, acc[t]);
    }
  }
  int n0 = lane & 15;
  int rO = (lane >> 4) * 8;
#pragma unroll
  for (int t = 0; t < 4; ++t) {
    int n = colBase + t * 16 + n0;
    float bn = bin[n];
#pragma unroll
    for (int j = 0; j < 8; ++j) {
      int r = rowBase + rO + j;
      float v = acc[t][j] + bn;
      encf[(size_t)r * MD + n] = v;
      encb[(size_t)r * MD + n] = f2bf(v);
    }
  }
}

// ---------- kernel 3: upd[d] = 0.1 * mean_b(sum_s enc) = 0.025 * colsum ----------

__global__ __launch_bounds__(256)
void k_colsum(const float* __restrict__ encf, float* __restrict__ upd) {
  __shared__ float red[256];
  int d = blockIdx.x;
  float s = 0.f;
  for (int r = threadIdx.x; r < ROWS; r += 256) s += encf[(size_t)r * MD + d];
  red[threadIdx.x] = s;
  __syncthreads();
  for (int off = 128; off > 0; off >>= 1) {
    if (threadIdx.x < off) red[threadIdx.x] += red[threadIdx.x + off];
    __syncthreads();
  }
  if (threadIdx.x == 0) upd[d] = 0.025f * red[0];
}

// ---------- kernel 4: new_memory = 0.9*memory + upd[d] ----------

__global__ __launch_bounds__(256)
void k_newmem(const float* __restrict__ mem, const float* __restrict__ upd,
              float* __restrict__ nm) {
  int idx = blockIdx.x * 256 + threadIdx.x;
  nm[idx] = 0.9f * mem[idx] + upd[idx & (MD - 1)];
}

// ---------- kernel 5: fused flash attention over memory bank ----------
// One block = 16 query rows, 4 waves. Chunk = 64 memory slots.
// Query block held entirely in registers (16 K-fragments = 128 VGPRs/wave).

__global__ __launch_bounds__(128)
void k_attn(const __bf16* __restrict__ encb, const __bf16* __restrict__ memb,
            const __bf16* __restrict__ memTb, __bf16* __restrict__ mob) {
  __shared__ float  sT[16 * 64];        // score tile
  __shared__ __bf16 pT[16 * 64];        // probs (bf16, A-operand for PV)
  __shared__ float  rowM[16], rowL[16], rowScl[16];

  int tid  = threadIdx.x;
  int lane = tid & 31;
  int wave = tid >> 5;
  int rowBase = blockIdx.x * 16;

  // entire 16x512 query block -> registers (A fragments for all 16 K-steps)
  v16bf aF[16];
#pragma unroll
  for (int kk = 0; kk < 16; ++kk)
    aF[kk] = load_frag_bf(encb, MD, rowBase, kk * 32, lane);

  if (tid < 16) { rowM[tid] = -3.0e38f; rowL[tid] = 0.f; }
  __syncthreads();

  v8f acc[8] = {};  // columns [wave*128, wave*128+128) of the output

  for (int chunk = 0; chunk < MS; chunk += 64) {
    // prefetch next chunk's K rows (L2-resident but cheap to hint)
    if (chunk + 64 < MS)
      __builtin_prefetch(memb + (size_t)(chunk + 64 + wave * 16 + (lane & 15)) * MD, 0, 1);

    // --- S tile: wave computes cols [chunk + wave*16, +16), K = 512 ---
    int colBase = chunk + wave * 16;
    v8f s = {};
#pragma unroll
    for (int kk = 0; kk < 16; ++kk) {
      v16bf b = load_frag_bf(memb, MD, colBase, kk * 32, lane);
      s = wmma_bf16(aF[kk], b, s);
    }
    {
      int n = wave * 16 + (lane & 15);
      int rO = (lane >> 4) * 8;
#pragma unroll
      for (int j = 0; j < 8; ++j) sT[(rO + j) * 64 + n] = s[j];
    }
    __syncthreads();

    // --- online softmax over this 64-wide chunk (one thread per row) ---
    if (tid < 16) {
      float mOld = rowM[tid];
      float mc = mOld;
#pragma unroll 8
      for (int c = 0; c < 64; ++c) mc = fmaxf(mc, sT[tid * 64 + c]);
      float scl = __expf(mOld - mc);      // 0 on first chunk (mOld ~ -inf)
      float sum = 0.f;
#pragma unroll 8
      for (int c = 0; c < 64; ++c) {
        float p = __expf(sT[tid * 64 + c] - mc);
        pT[tid * 64 + c] = f2bf(p);
        sum += p;
      }
      rowL[tid] = rowL[tid] * scl + sum;
      rowM[tid] = mc;
      rowScl[tid] = scl;
    }
    __syncthreads();

    // --- rescale accumulator, then acc += P(16x64) @ memChunk(64x512) ---
    int rO = (lane >> 4) * 8;
    float scl[8];
#pragma unroll
    for (int j = 0; j < 8; ++j) scl[j] = rowScl[rO + j];
#pragma unroll
    for (int t = 0; t < 8; ++t)
#pragma unroll
      for (int j = 0; j < 8; ++j) acc[t][j] *= scl[j];

    v16bf pa0 = load_frag_bf(pT, 64, 0, 0, lane);
    v16bf pa1 = load_frag_bf(pT, 64, 0, 32, lane);
#pragma unroll
    for (int t = 0; t < 8; ++t) {
      int dIdx = wave * 128 + t * 16;                 // output column tile
      v16bf b0 = load_frag_bf(memTb, MS, dIdx, chunk, lane);
      v16bf b1 = load_frag_bf(memTb, MS, dIdx, chunk + 32, lane);
      acc[t] = wmma_bf16(pa0, b0, acc[t]);
      acc[t] = wmma_bf16(pa1, b1, acc[t]);
    }
    __syncthreads();  // protect sT/pT before next chunk
  }

  // --- normalize by row sums and emit bf16 mem_out ---
  int rO = (lane >> 4) * 8;
  float li[8];
#pragma unroll
  for (int j = 0; j < 8; ++j) li[j] = 1.0f / rowL[rO + j];
  int n0 = lane & 15;
#pragma unroll
  for (int t = 0; t < 8; ++t) {
    int n = wave * 128 + t * 16 + n0;
#pragma unroll
    for (int j = 0; j < 8; ++j)
      mob[(size_t)(rowBase + rO + j) * MD + n] = f2bf(acc[t][j] * li[j]);
  }
}

// ---------- kernel 6: out = mem_out @ W_out^T + b_out (8192x1024, K=512) ----------

__global__ __launch_bounds__(256)
void k_decode(const __bf16* __restrict__ mob, const __bf16* __restrict__ Woutb,
              const float* __restrict__ bout, float* __restrict__ out) {
  int lane = threadIdx.x & 31;
  int wave = threadIdx.x >> 5;
  int rowBase = blockIdx.x * 64 + (wave & 3) * 16;
  int colBase = blockIdx.y * 128 + (wave >> 2) * 64;
  v8f acc[4] = {};
  for (int k = 0; k < MD; k += 32) {
    v16bf a = load_frag_bf(mob, MD, rowBase, k, lane);
#pragma unroll
    for (int t = 0; t < 4; ++t) {
      v16bf b = load_frag_bf(Woutb, MD, colBase + 16 * t, k, lane);
      acc[t] = wmma_bf16(a, b, acc[t]);
    }
  }
  int n0 = lane & 15;
  int rO = (lane >> 4) * 8;
#pragma unroll
  for (int t = 0; t < 4; ++t) {
    int n = colBase + t * 16 + n0;
    float bn = bout[n];
#pragma unroll
    for (int j = 0; j < 8; ++j)
      out[(size_t)(rowBase + rO + j) * IND + n] = acc[t][j] + bn;
  }
}

// ---------- launch ----------

extern "C" void kernel_launch(void* const* d_in, const int* in_sizes, int n_in,
                              void* d_out, int out_size, void* d_ws, size_t ws_size,
                              hipStream_t stream) {
  const float* x      = (const float*)d_in[0];
  const float* Win    = (const float*)d_in[1];
  const float* bin    = (const float*)d_in[2];
  const float* Wout   = (const float*)d_in[3];
  const float* bout   = (const float*)d_in[4];
  // d_in[5]=W_write, d_in[6]=b_write: softmax over a size-1 axis == 1.0, unused.
  const float* memory = (const float*)d_in[7];

  float* out     = (float*)d_out;                       // [8192,1024]
  float* new_mem = out + (size_t)ROWS * IND;            // [8192,512]

  char* w = (char*)d_ws;
  size_t off = 0;
  float*  enc_f  = (float*)(w + off);  off += (size_t)ROWS * MD * 4;   // 16 MB
  __bf16* enc_b  = (__bf16*)(w + off); off += (size_t)ROWS * MD * 2;   //  8 MB
  __bf16* mem_b  = (__bf16*)(w + off); off += (size_t)MS * MD * 2;     //  8 MB
  __bf16* memT_b = (__bf16*)(w + off); off += (size_t)MS * MD * 2;     //  8 MB
  __bf16* mo_b   = (__bf16*)(w + off); off += (size_t)ROWS * MD * 2;   //  8 MB
  __bf16* x_b    = (__bf16*)(w + off); off += (size_t)ROWS * IND * 2;  // 16 MB
  __bf16* Win_b  = (__bf16*)(w + off); off += (size_t)MD * IND * 2;    //  1 MB
  __bf16* Wout_b = (__bf16*)(w + off); off += (size_t)IND * MD * 2;    //  1 MB
  float*  upd    = (float*)(w + off);                                  //  2 KB

  // one-shot precision conversions (bandwidth-bound, ~2 us total)
  k_cvt_bf16<<<dim3((ROWS * IND / 8) / 256), dim3(256), 0, stream>>>(x, x_b, ROWS * IND / 8);
  k_cvt_bf16<<<dim3((MD * IND / 8) / 256), dim3(256), 0, stream>>>(Win, Win_b, MD * IND / 8);
  k_cvt_bf16<<<dim3((IND * MD / 8) / 256), dim3(256), 0, stream>>>(Wout, Wout_b, IND * MD / 8);
  k_memconv<<<dim3((MS * MD) / 256), dim3(256), 0, stream>>>(memory, mem_b, memT_b);

  k_encode <<<dim3(ROWS / 64, MD / 128), dim3(256), 0, stream>>>(x_b, Win_b, bin, enc_f, enc_b);
  k_colsum <<<dim3(MD), dim3(256), 0, stream>>>(enc_f, upd);
  k_newmem <<<dim3((MS * MD) / 256), dim3(256), 0, stream>>>(memory, upd, new_mem);
  k_attn   <<<dim3(ROWS / 16), dim3(128), 0, stream>>>(enc_b, mem_b, memT_b, mo_b);
  k_decode <<<dim3(ROWS / 64, IND / 128), dim3(256), 0, stream>>>(mo_b, Wout_b, bout, out);
}
